// RelGraphConvLayer_17592186044975
// MI455X (gfx1250) — compile-verified, matrix operand order checked by reading, not run on hardware
//
#include <hip/hip_runtime.h>

#define NNODES 100000
#define RREL   4
#define EDGES  160000
#define DIN    128
#define DOUT   128

typedef __attribute__((ext_vector_type(2)))  float  v2f;
typedef __attribute__((ext_vector_type(8)))  float  v8f;
typedef __attribute__((ext_vector_type(16))) __bf16 v16bf;

// ---------------------------------------------------------------------------
// 1) in-degree per (relation, dst) : deg[r*N + d] += 1.0f
// ---------------------------------------------------------------------------
__global__ __launch_bounds__(256) void deg_kernel(const int* __restrict__ dst,
                                                  float* __restrict__ deg) {
    int i = blockIdx.x * blockDim.x + threadIdx.x;   // over R*E
    if (i >= RREL * EDGES) return;
    int r = i / EDGES;
    int d = dst[i];
    atomicAdd(&deg[(size_t)r * NNODES + d], 1.0f);
}

// ---------------------------------------------------------------------------
// 2) edge scatter in FEATURE space: a[r][dst] += x[src] / clamp(deg,1)
//    one wave32 per edge; each lane moves a float4 chunk (32*16B = 512B row).
//    x (51MB) is L2-resident on MI455X (192MB L2) -> random gathers hit L2.
// ---------------------------------------------------------------------------
__global__ __launch_bounds__(256) void scatter_kernel(const float* __restrict__ x,
                                                      const int* __restrict__ src,
                                                      const int* __restrict__ dst,
                                                      const float* __restrict__ deg,
                                                      float* __restrict__ a) {
    unsigned gw = (blockIdx.x * blockDim.x + threadIdx.x) >> 5;   // edge id
    if (gw >= RREL * EDGES) return;
    int lane = threadIdx.x & 31;
    int r = gw / EDGES;
    int s = src[gw];
    int d = dst[gw];
    float inv = 1.0f / fmaxf(deg[(size_t)r * NNODES + d], 1.0f);
    const float4 v = *reinterpret_cast<const float4*>(x + (size_t)s * DIN + lane * 4);
    float* o = a + ((size_t)r * NNODES + d) * DIN + lane * 4;
    atomicAdd(o + 0, v.x * inv);
    atomicAdd(o + 1, v.y * inv);
    atomicAdd(o + 2, v.z * inv);
    atomicAdd(o + 3, v.w * inv);
}

// ---------------------------------------------------------------------------
// 3) fused GEMM:  out = relu( [a0|a1|a2|a3|x] @ [W0;W1;W2;W3;loopW] + bias )
//    block = one 16-row tile; 8 waves = 8 col tiles of 16; K = 5*128 = 640.
//    Wave32 WMMA: D(16x16,f32) = A(16x4,f32) x B(4x16,f32) + C, 160 per tile.
// ---------------------------------------------------------------------------
__global__ __launch_bounds__(256) void gemm_kernel(const float* __restrict__ x,
                                                   const float* __restrict__ weight,
                                                   const float* __restrict__ loop_weight,
                                                   const float* __restrict__ bias,
                                                   const float* __restrict__ a,
                                                   float* __restrict__ out) {
    const int row0 = blockIdx.x * 16;
    const int wv   = threadIdx.x >> 5;      // 0..7  -> column tile
    const int lane = threadIdx.x & 31;
    const int m    = lane & 15;             // M (A) / N (B,C) index within tile
    const int hi   = lane >> 4;             // half-wave select
    const int col  = wv * 16 + m;

    v8f acc = {};

#if __has_builtin(__builtin_amdgcn_wmma_f32_16x16x4_f32)
    // A 16x4 f32: lanes 0-15 -> K={0,1}, lanes 16-31 -> K={2,3} (v2f per lane)
    // B 4x16 f32: same split on K; N = lane&15
    const int koff = hi * 2;
    for (int seg = 0; seg < 5; ++seg) {
        const float* Ab = (seg < 4) ? (a + (size_t)seg * NNODES * DIN) : x;
        const float* Bb = (seg < 4) ? (weight + (size_t)seg * DIN * DOUT) : loop_weight;
        const float* arow = Ab + (size_t)(row0 + m) * DIN;
        #pragma unroll 4
        for (int k = 0; k < DIN; k += 4) {
            v2f av, bv;
            av.x = arow[k + koff];
            av.y = arow[k + koff + 1];
            bv.x = Bb[(size_t)(k + koff) * DOUT + col];
            bv.y = Bb[(size_t)(k + koff + 1) * DOUT + col];
            acc = __builtin_amdgcn_wmma_f32_16x16x4_f32(
                false, av, false, bv, (short)0, acc, false, false);
        }
    }
#else
    // Fallback: codegen-confirmed bf16 WMMA (f32 accumulate), K-step 32.
    // 16-bit A 16x32 layout: lanes<16 hold K=0..7 & 16..23, lanes>=16 +8.
    const int koff = hi * 8;
    for (int seg = 0; seg < 5; ++seg) {
        const float* Ab = (seg < 4) ? (a + (size_t)seg * NNODES * DIN) : x;
        const float* Bb = (seg < 4) ? (weight + (size_t)seg * DIN * DOUT) : loop_weight;
        const float* arow = Ab + (size_t)(row0 + m) * DIN;
        for (int k = 0; k < DIN; k += 32) {
            v16bf av, bv;
            #pragma unroll
            for (int i = 0; i < 8; ++i) {
                av[i]     = (__bf16)arow[k + koff + i];
                av[8 + i] = (__bf16)arow[k + koff + 16 + i];
                bv[i]     = (__bf16)Bb[(size_t)(k + koff + i) * DOUT + col];
                bv[8 + i] = (__bf16)Bb[(size_t)(k + koff + 16 + i) * DOUT + col];
            }
            acc = __builtin_amdgcn_wmma_f32_16x16x32_bf16(
                false, av, false, bv, (short)0, acc, false, false);
        }
    }
#endif

    // C/D 16x16 f32 layout: VGPR v -> row M = v + hi*8, N = lane&15
    const float bcol = bias[col];
    #pragma unroll
    for (int v = 0; v < 8; ++v) {
        int row = row0 + v + hi * 8;
        float val = acc[v] + bcol;
        out[(size_t)row * DOUT + col] = val > 0.0f ? val : 0.0f;
    }
}

// ---------------------------------------------------------------------------
// launch
// ---------------------------------------------------------------------------
extern "C" void kernel_launch(void* const* d_in, const int* in_sizes, int n_in,
                              void* d_out, int out_size, void* d_ws, size_t ws_size,
                              hipStream_t stream) {
    const float* x           = (const float*)d_in[0];
    const float* weight      = (const float*)d_in[1];
    const float* loop_weight = (const float*)d_in[2];
    const float* h_bias      = (const float*)d_in[3];
    const int*   src_idx     = (const int*)d_in[4];
    const int*   dst_idx     = (const int*)d_in[5];
    float*       out         = (float*)d_out;

    // workspace layout: deg[R*N] f32, then a[R*N*128] f32 (~206.4 MB)
    float* deg = (float*)d_ws;
    float* a   = deg + (size_t)RREL * NNODES;
    size_t zbytes = ((size_t)RREL * NNODES + (size_t)RREL * NNODES * DIN) * sizeof(float);
    hipMemsetAsync(d_ws, 0, zbytes, stream);

    {
        int total = RREL * EDGES;
        deg_kernel<<<(total + 255) / 256, 256, 0, stream>>>(dst_idx, deg);
    }
    {
        long long threads = (long long)RREL * EDGES * 32;   // one wave per edge
        int blocks = (int)((threads + 255) / 256);          // 80000
        scatter_kernel<<<blocks, 256, 0, stream>>>(x, src_idx, dst_idx, deg, a);
    }
    gemm_kernel<<<NNODES / 16, 256, 0, stream>>>(x, weight, loop_weight, h_bias, a, out);
}
